// LabOutcome_41635412967896
// MI455X (gfx1250) — compile-verified, matrix operand order checked by reading, not run on hardware
//
#include <hip/hip_runtime.h>
#include <math.h>

typedef __attribute__((ext_vector_type(16))) _Float16 v16h;
typedef __attribute__((ext_vector_type(8)))  float    v8f;

#define B_   256
#define T_   512
#define F_   500
#define L_   10
#define FPL_ 50

// One block (1024 threads = 32 waves) per batch element.
// Wave w owns features [16w, 16w+16). Lane's column n = lane&15 -> feature f0+n.
// WMMA: D = ones(16x32,f16) x B(32x16,f16) + C  ==> every row of D = column sums
// of B, i.e. the partial sum over 32 t-steps of diff at that lane's feature.
// Columns with f >= 500 compute garbage (clamped address) but are never read.
__global__ __launch_bounds__(1024) void labvar_main(
    const float* __restrict__ Xprev, const float* __restrict__ Xpost,
    const float* __restrict__ t_in,  const float* __restrict__ tup,
    const float* __restrict__ tlow,  const float* __restrict__ costs,
    float* __restrict__ out, float* __restrict__ partials)
{
    __shared__ float absS[F_];
    __shared__ float labd[L_], labb[L_], labc[L_];

    const int b    = blockIdx.x;
    const int tid  = threadIdx.x;
    const int wave = tid >> 5;          // 0..31 : feature group
    const int lane = tid & 31;
    const int n    = lane & 15;         // column within 16x16 tile
    const int kb   = (lane & 16);       // lanes 16-31 hold K=16..31 rows
    const int f    = wave * 16 + n;
    const bool valid = (f < F_);
    const int fc   = valid ? f : (F_ - 1);   // clamp address; result discarded

    v16h ones;
#pragma unroll
    for (int i = 0; i < 16; ++i) ones[i] = (_Float16)1.0f;

    v8f acc = {};
    const float* basep = Xprev + ((size_t)b * T_) * F_ + fc;
    const float* baseq = Xpost + ((size_t)b * T_) * F_ + fc;

    for (int t0 = 0; t0 < T_; t0 += 32) {
        const float* pp = basep + (size_t)(t0 + kb) * F_;
        const float* pq = baseq + (size_t)(t0 + kb) * F_;
        v16h bm;
#pragma unroll
        for (int j = 0; j < 16; ++j) {
            // pure stream, no reuse -> non-temporal hint
            float d = __builtin_nontemporal_load(pp) -
                      __builtin_nontemporal_load(pq);
            bm[j] = (_Float16)d;
            pp += F_;
            pq += F_;
        }
        // codegen-confirmed: v_wmma_f32_16x16x32_f16
        acc = __builtin_amdgcn_wmma_f32_16x16x32_f16(
                  false, ones, false, bm, (short)0, acc, false, false);
    }

    // acc[0] = D[row0 (or row8), n] = sum over all T of diff at feature f
    if (lane < 16 && valid) absS[f] = fabsf(acc[0]) * (1.0f / (float)T_);
    __syncthreads();

    if (tid < L_) {
        const int l = tid;
        float v = 0.0f;
#pragma unroll
        for (int i = 0; i < FPL_; ++i) v += absS[l * FPL_ + i];
        const float tv   = t_in[b * L_ + l];
        const float to   = 1.0f / (1.0f + expf(-(tv - 0.5f) * 1000.0f));
        const float gate = (tv > 0.5f) ? 1.0f : 0.0f;
        const float tu   = tup[b * L_ + l];
        const float tl   = tlow[b * L_ + l];
        labd[l] = v * gate * to;                       // deltats * thisorder
        labb[l] = (tu == tl) ? fabsf(tv - tl) : 0.0f;  // bound penalty
        labc[l] = to * costs[l];                       // ordering cost
    }
    __syncthreads();

    if (tid == 0) {
        float ev = 0.0f, eb = 0.0f, cc = 0.0f;
        for (int l = 0; l < L_; ++l) { ev += labd[l]; eb += labb[l]; cc += labc[l]; }
        out[b] = ev * 1.0f - eb * 1.0f - cc * 1.0f;    // WDX=WLB=WC=1
        partials[b]          = ev;
        partials[B_ + b]     = eb;
        partials[2 * B_ + b] = cc;
    }
}

// Tiny second pass: means over B for the three diagnostics.
__global__ __launch_bounds__(256) void labvar_reduce(
    const float* __restrict__ partials, float* __restrict__ out)
{
    __shared__ float s0[B_], s1[B_], s2[B_];
    const int tid = threadIdx.x;
    s0[tid] = partials[tid];
    s1[tid] = partials[B_ + tid];
    s2[tid] = partials[2 * B_ + tid];
    __syncthreads();
    for (int off = B_ / 2; off > 0; off >>= 1) {
        if (tid < off) {
            s0[tid] += s0[tid + off];
            s1[tid] += s1[tid + off];
            s2[tid] += s2[tid + off];
        }
        __syncthreads();
    }
    if (tid == 0) {
        out[B_ + 0] = s0[0] * (1.0f / (float)B_);  // evm
        out[B_ + 1] = s1[0] * (1.0f / (float)B_);  // ebm
        out[B_ + 2] = s2[0] * (1.0f / (float)B_);  // cm
    }
}

extern "C" void kernel_launch(void* const* d_in, const int* in_sizes, int n_in,
                              void* d_out, int out_size, void* d_ws, size_t ws_size,
                              hipStream_t stream) {
    const float* Xprev = (const float*)d_in[0];   // [B,T,F]
    const float* Xpost = (const float*)d_in[1];   // [B,T,F]
    const float* t_in  = (const float*)d_in[2];   // [B,L]
    const float* tup   = (const float*)d_in[3];   // [B,L]
    const float* tlow  = (const float*)d_in[4];   // [B,L]
    const float* costs = (const float*)d_in[5];   // [L]
    float* out = (float*)d_out;                   // [B + 3]
    float* partials = (float*)d_ws;               // 3*B floats (3 KB)

    labvar_main<<<B_, 1024, 0, stream>>>(Xprev, Xpost, t_in, tup, tlow, costs,
                                         out, partials);
    labvar_reduce<<<1, 256, 0, stream>>>(partials, out);
}
